// SelfAttention_78211354460866
// MI455X (gfx1250) — compile-verified, matrix operand order checked by reading, not run on hardware
//
#include <hip/hip_runtime.h>

typedef __bf16 bf16;
typedef __attribute__((ext_vector_type(16))) __bf16 v16bf;
typedef __attribute__((ext_vector_type(8)))  __bf16 v8bf;
typedef __attribute__((ext_vector_type(8)))  float  v8f;
typedef unsigned int uint32x4 __attribute__((ext_vector_type(4)));
typedef int          int32x4  __attribute__((ext_vector_type(4)));
typedef int          int32x8  __attribute__((ext_vector_type(8)));

constexpr int kS  = 2048;
constexpr int kC  = 3072;
constexpr int kH  = 24;
constexpr int kHD = 128;
constexpr int kChunk = 1024;                 // K elements staged per TDM transfer
constexpr int kNChunk = kC / kChunk;         // 3

// ---------------------------------------------------------------------------
// WMMA helper (CDNA5 wave32, 16x16x32 bf16 -> f32)
// ---------------------------------------------------------------------------
__device__ __forceinline__ v8f wmma_bf16(v16bf a, v16bf b, v8f c) {
  return __builtin_amdgcn_wmma_f32_16x16x32_bf16(
      /*neg_a=*/false, a, /*neg_b=*/false, b,
      /*c_mod=*/(short)0, c, /*reuse_a=*/false, /*reuse_b=*/false);
}

// Load a 16x32 A-fragment (or B-fragment via B^T) from row-major bf16 memory.
// Lane L (0..15):  row = row0+L,    K = {k0..k0+7, k0+16..k0+23}
// Lane L (16..31): row = row0+L-16, K = {k0+8..k0+15, k0+24..k0+31}
__device__ __forceinline__ v16bf load_frag(const bf16* __restrict__ base,
                                           int row0, int ld, int k0) {
  int lane = threadIdx.x & 31;
  int m    = lane & 15;
  int klo  = (lane & 16) ? 8 : 0;
  const bf16* p = base + (size_t)(row0 + m) * (size_t)ld + (size_t)(k0 + klo);
  v8bf lo = *(const v8bf*)(p);
  v8bf hi = *(const v8bf*)(p + 16);
  v16bf a;
#pragma unroll
  for (int i = 0; i < 8; ++i) { a[i] = lo[i]; a[i + 8] = hi[i]; }
  return a;
}

// ---------------------------------------------------------------------------
// TDM: stage a 16-row x kChunk bf16 panel (row stride kC) into LDS.
// D# layout per CDNA5 ISA.
// ---------------------------------------------------------------------------
__device__ __forceinline__ unsigned lds_offset_of(void* p) {
  return (unsigned)(unsigned long long)(__attribute__((address_space(3))) char*)p;
}

__device__ __forceinline__ void tdm_load_panel(const bf16* gsrc, unsigned lds_off) {
  unsigned long long ga = (unsigned long long)gsrc;
  uint32x4 g0;
  g0.x = 1u;                                            // count=1, user mode
  g0.y = lds_off;                                       // lds_addr
  g0.z = (unsigned)(ga & 0xFFFFFFFFu);                  // global_addr[31:0]
  g0.w = (unsigned)((ga >> 32) & 0x1FFFFFFu) | (2u << 30); // addr[56:32] | type=2
  int32x8 g1;
  g1[0] = 0x00010000;                                   // data_size = 2 bytes
  g1[1] = (kChunk & 0xFFFF) << 16;                      // tensor_dim0[15:0]
  g1[2] = (kChunk >> 16) | (16 << 16);                  // tensor_dim0[31:16] | tensor_dim1[15:0]
  g1[3] = (kChunk & 0xFFFF) << 16;                      // tile_dim0 = kChunk
  g1[4] = 16;                                           // tile_dim1 = 16 rows
  g1[5] = kC;                                           // tensor_dim0_stride[31:0]
  g1[6] = 0;
  g1[7] = 0;
  int32x4 z4 = {0, 0, 0, 0};
#if defined(__clang_major__) && __clang_major__ >= 23
  int32x8 z8 = {0, 0, 0, 0, 0, 0, 0, 0};
  __builtin_amdgcn_tensor_load_to_lds(g0, g1, z4, z4, z8, 0);
#else
  __builtin_amdgcn_tensor_load_to_lds(g0, g1, z4, z4, 0);
#endif
}

// ---------------------------------------------------------------------------
// fp32 -> bf16 cast
// ---------------------------------------------------------------------------
__global__ void __launch_bounds__(256) cast_bf16_kernel(
    const float* __restrict__ in, bf16* __restrict__ out, int n) {
  int i = blockIdx.x * 256 + threadIdx.x;
  if (i < n) out[i] = (bf16)in[i];
}

// ---------------------------------------------------------------------------
// GEMM: D[kS,kC] = A[kS,kC](bf16) @ Bt[kC,kC](bf16)^T + bias, fp32 out.
// Block = 8 waves sharing one 16-row A panel (TDM-staged, double-buffered LDS);
// each wave owns a 16x64 tile. B fragments ping-pong between two register
// sets (unroll-by-2) so no cross-buffer copies are needed.
// ---------------------------------------------------------------------------
__global__ void __launch_bounds__(256) gemm_bf16_kernel(
    const bf16* __restrict__ A, const bf16* __restrict__ Bt,
    const float* __restrict__ bias, float* __restrict__ D) {
  __shared__ __align__(16) bf16 Atile[2][16 * kChunk];   // 2 x 32 KB

  int lane = threadIdx.x & 31;
  int wave = threadIdx.x >> 5;
  int m0 = blockIdx.y * 16;
  int n0 = (blockIdx.x * 8 + wave) * 64;

  if (wave == 0 && lane == 0)
    tdm_load_panel(A + (size_t)m0 * kC, lds_offset_of(&Atile[0][0]));

  v8f acc[4] = {};
  for (int c = 0; c < kNChunk; ++c) {
    if (wave == 0 && lane == 0) {
      if (c + 1 < kNChunk) {
        tdm_load_panel(A + (size_t)m0 * kC + (c + 1) * kChunk,
                       lds_offset_of(&Atile[(c + 1) & 1][0]));
        __builtin_amdgcn_s_wait_tensorcnt(1);   // chunk c complete, c+1 in flight
      } else {
        __builtin_amdgcn_s_wait_tensorcnt(0);   // drain: final chunk complete
      }
    }
    __syncthreads();

    const bf16* apanel = &Atile[c & 1][0];
    int kbase = c * kChunk;

    v16bf bA[4], bB[4];
#pragma unroll
    for (int t = 0; t < 4; ++t)
      bA[t] = load_frag(Bt, n0 + t * 16, kC, kbase);

    for (int kk = 0; kk < kChunk; kk += 64) {
      // prefetch hint on the B stream
      __builtin_prefetch(Bt + (size_t)(n0 + (lane & 15)) * kC + kbase + kk + 128, 0, 1);
      // fill the other register set while WMMAs consume this one (ping-pong)
#pragma unroll
      for (int t = 0; t < 4; ++t)
        bB[t] = load_frag(Bt, n0 + t * 16, kC, kbase + kk + 32);
      v16bf a0 = load_frag(apanel, 0, kChunk, kk);          // ds_load from LDS
#pragma unroll
      for (int t = 0; t < 4; ++t) acc[t] = wmma_bf16(a0, bA[t], acc[t]);
#pragma unroll
      for (int t = 0; t < 4; ++t)
        bA[t] = load_frag(Bt, n0 + t * 16, kC, kbase + kk + 64); // last iter
                                                                 // overreads into ws slack
      v16bf a1 = load_frag(apanel, 0, kChunk, kk + 32);
#pragma unroll
      for (int t = 0; t < 4; ++t) acc[t] = wmma_bf16(a1, bB[t], acc[t]);
    }
    __syncthreads();   // all waves done with this buffer before TDM reuses it
  }

  int nlo   = lane & 15;
  int mhalf = (lane & 16) ? 8 : 0;
#pragma unroll
  for (int t = 0; t < 4; ++t) {
    int col = n0 + t * 16 + nlo;
    float bia = bias ? bias[col] : 0.0f;
#pragma unroll
    for (int r = 0; r < 8; ++r) {
      int row = m0 + r + mhalf;
      D[(size_t)row * kC + col] = acc[t][r] + bia;
    }
  }
}

// ---------------------------------------------------------------------------
// RMSNorm (over HD) + RoPE, fp32 [S,C] -> bf16 head-major [H][S][HD].
// One wave per (s, h); each lane owns 4 consecutive dims (2 rope pairs).
// ---------------------------------------------------------------------------
__global__ void __launch_bounds__(128) rmsnorm_rope_kernel(
    const float* __restrict__ in, const float* __restrict__ g,
    const float* __restrict__ freqs, bf16* __restrict__ out) {
  int lane = threadIdx.x & 31;
  int wid  = blockIdx.x * 4 + (threadIdx.x >> 5);
  int s = wid / kH;
  int h = wid % kH;
  const float* row = in + (size_t)s * kC + h * kHD;
  int d0 = lane * 4;
  float v0 = row[d0], v1 = row[d0 + 1], v2 = row[d0 + 2], v3 = row[d0 + 3];
  float ss = v0 * v0 + v1 * v1 + v2 * v2 + v3 * v3;
#pragma unroll
  for (int off = 16; off > 0; off >>= 1) ss += __shfl_xor(ss, off, 32);
  float r = rsqrtf(ss * (1.0f / kHD) + 1e-6f);
  v0 *= r * g[d0];     v1 *= r * g[d0 + 1];
  v2 *= r * g[d0 + 2]; v3 *= r * g[d0 + 3];
  // freqs: [S, HD/2, 2, 2]; out[p,i] = f[p,i,0]*h[2p] + f[p,i,1]*h[2p+1]
  const float* f = freqs + (size_t)s * (kHD / 2) * 4 + (size_t)(d0 / 2) * 4;
  float o0 = f[0] * v0 + f[1] * v1;
  float o1 = f[2] * v0 + f[3] * v1;
  float o2 = f[4] * v2 + f[5] * v3;
  float o3 = f[6] * v2 + f[7] * v3;
  bf16* op = out + ((size_t)h * kS + s) * kHD + d0;
  op[0] = (bf16)o0; op[1] = (bf16)o1; op[2] = (bf16)o2; op[3] = (bf16)o3;
}

// ---------------------------------------------------------------------------
// V: fp32 [S,C] -> bf16 transposed [H][HD][S]
// ---------------------------------------------------------------------------
__global__ void __launch_bounds__(256) vtrans_kernel(
    const float* __restrict__ vf, bf16* __restrict__ vt) {
  size_t t = (size_t)blockIdx.x * 256 + threadIdx.x;   // t = hd*S + s
  int s  = (int)(t & (kS - 1));
  int hd = (int)(t >> 11);                              // S = 2048 = 2^11
  int h = hd >> 7, d = hd & (kHD - 1);
  vt[t] = (bf16)vf[(size_t)s * kC + h * kHD + d];
}

// ---------------------------------------------------------------------------
// Flash attention: one wave per (head, 16-query block). Online softmax over
// 32-key blocks; P tile staged through per-wave LDS to convert C-layout to
// A-layout for the PV WMMAs. Output: bf16 attn[S][C] (col = h*HD + d).
// ---------------------------------------------------------------------------
#define ATT_WAVES 4
__global__ void __launch_bounds__(ATT_WAVES * 32) attn_kernel(
    const bf16* __restrict__ qb, const bf16* __restrict__ kb,
    const bf16* __restrict__ vt, bf16* __restrict__ attnb) {
  __shared__ __align__(16) bf16 plds[ATT_WAVES][16 * 32];
  int lane = threadIdx.x & 31;
  int wave = threadIdx.x >> 5;
  int h  = blockIdx.y;
  int m0 = (blockIdx.x * ATT_WAVES + wave) * 16;

  const bf16* qh = qb + (size_t)h * kS * kHD;
  const bf16* kh = kb + (size_t)h * kS * kHD;
  const bf16* vh = vt + (size_t)h * kHD * kS;

  v16bf aq[4];
#pragma unroll
  for (int kk = 0; kk < 4; ++kk) aq[kk] = load_frag(qh, m0, kHD, kk * 32);

  v8f acc[8] = {};
  float mrow[8], lrow[8];
#pragma unroll
  for (int r = 0; r < 8; ++r) { mrow[r] = -3.0e38f; lrow[r] = 0.0f; }

  const float scale = 0.088388347648318447f;   // 1/sqrt(128)
  int nlo   = lane & 15;
  int mhalf = (lane & 16) ? 8 : 0;
  bf16* pw = &plds[wave][0];

  for (int j0 = 0; j0 < kS; j0 += 32) {
    // ---- scores: 16x32 tile, K = HD = 128 ----
    v8f s0 = {}, s1 = {};
#pragma unroll
    for (int kk = 0; kk < 4; ++kk) {
      v16bf b0 = load_frag(kh, j0, kHD, kk * 32);
      s0 = wmma_bf16(aq[kk], b0, s0);
      v16bf b1 = load_frag(kh, j0 + 16, kHD, kk * 32);
      s1 = wmma_bf16(aq[kk], b1, s1);
    }
    // ---- online softmax; row stats per lane-half (width-16 shuffles) ----
    float alpha[8];
#pragma unroll
    for (int r = 0; r < 8; ++r) {
      float x0 = s0[r] * scale, x1 = s1[r] * scale;
      float vmax = fmaxf(x0, x1);
#pragma unroll
      for (int off = 1; off < 16; off <<= 1)
        vmax = fmaxf(vmax, __shfl_xor(vmax, off, 16));
      float mn = fmaxf(mrow[r], vmax);
      float a  = __expf(mrow[r] - mn);
      mrow[r] = mn;
      float p0 = __expf(x0 - mn), p1 = __expf(x1 - mn);
      float rs = p0 + p1;
#pragma unroll
      for (int off = 1; off < 16; off <<= 1) rs += __shfl_xor(rs, off, 16);
      lrow[r] = lrow[r] * a + rs;
      alpha[r] = a;
      int mm = r + mhalf;                    // C-layout row for this lane half
      pw[mm * 32 + nlo]      = (bf16)p0;
      pw[mm * 32 + 16 + nlo] = (bf16)p1;
    }
#pragma unroll
    for (int t = 0; t < 8; ++t)
#pragma unroll
      for (int r = 0; r < 8; ++r) acc[t][r] *= alpha[r];

    // ---- re-read P as a 16x32 A-fragment (ds_load_b128; per-wave region,
    //      DS ops are in-order within a wave so no barrier needed) ----
    v16bf pa;
    {
      int klo = (lane & 16) ? 8 : 0;
      const bf16* pr = pw + nlo * 32 + klo;
      v8bf lo = *(const v8bf*)(pr);
      v8bf hi = *(const v8bf*)(pr + 16);
#pragma unroll
      for (int i = 0; i < 8; ++i) { pa[i] = lo[i]; pa[i + 8] = hi[i]; }
    }
    // ---- PV: 8 N-tiles of the 16x128 output ----
#pragma unroll
    for (int t = 0; t < 8; ++t) {
      v16bf bfrag = load_frag(vh, t * 16, kS, j0);   // B^T rows = V^T rows = dims
      acc[t] = wmma_bf16(pa, bfrag, acc[t]);
    }
  }

  float inv[8];
#pragma unroll
  for (int r = 0; r < 8; ++r) inv[r] = 1.0f / lrow[r];
#pragma unroll
  for (int t = 0; t < 8; ++t) {
    int col = h * kHD + t * 16 + nlo;
#pragma unroll
    for (int r = 0; r < 8; ++r) {
      int row = m0 + r + mhalf;
      attnb[(size_t)row * kC + col] = (bf16)(acc[t][r] * inv[r]);
    }
  }
}

// ---------------------------------------------------------------------------
// Orchestration
// ---------------------------------------------------------------------------
extern "C" void kernel_launch(void* const* d_in, const int* in_sizes, int n_in,
                              void* d_out, int out_size, void* d_ws, size_t ws_size,
                              hipStream_t stream) {
  const float* x     = (const float*)d_in[0];
  const float* freqs = (const float*)d_in[1];
  const float* wq    = (const float*)d_in[2];
  const float* bq    = (const float*)d_in[3];
  const float* wk    = (const float*)d_in[4];
  const float* bk    = (const float*)d_in[5];
  const float* wv    = (const float*)d_in[6];
  const float* bvp   = (const float*)d_in[7];
  const float* gq    = (const float*)d_in[8];
  const float* gk    = (const float*)d_in[9];
  const float* wo    = (const float*)d_in[10];
  const float* bo    = (const float*)d_in[11];

  char* ws = (char*)d_ws;
  size_t off = 0;
  auto take = [&](size_t bytes) -> char* {
    char* p = ws + off;
    off += (bytes + 255) & ~(size_t)255;
    return p;
  };
  const size_t SC = (size_t)kS * kC;
  const size_t CC = (size_t)kC * kC;
  bf16*  xb    = (bf16*)take(SC * 2);
  bf16*  wqb   = (bf16*)take(CC * 2);
  bf16*  wkb   = (bf16*)take(CC * 2);
  bf16*  wvb   = (bf16*)take(CC * 2);
  bf16*  wob   = (bf16*)take(CC * 2);
  float* qf    = (float*)take(SC * 4);
  float* kf    = (float*)take(SC * 4);
  float* vf    = (float*)take(SC * 4);
  bf16*  qbn   = (bf16*)take(SC * 2);
  bf16*  kbn   = (bf16*)take(SC * 2);
  bf16*  vtb   = (bf16*)take(SC * 2);
  bf16*  attnb = (bf16*)take(SC * 2);

  // 1) casts
  cast_bf16_kernel<<<(int)((SC + 255) / 256), 256, 0, stream>>>(x,  xb,  (int)SC);
  cast_bf16_kernel<<<(int)((CC + 255) / 256), 256, 0, stream>>>(wq, wqb, (int)CC);
  cast_bf16_kernel<<<(int)((CC + 255) / 256), 256, 0, stream>>>(wk, wkb, (int)CC);
  cast_bf16_kernel<<<(int)((CC + 255) / 256), 256, 0, stream>>>(wv, wvb, (int)CC);
  cast_bf16_kernel<<<(int)((CC + 255) / 256), 256, 0, stream>>>(wo, wob, (int)CC);

  // 2) q/k/v projections (TDM-staged A panel)
  dim3 gg(kC / 512, kS / 16);
  gemm_bf16_kernel<<<gg, 256, 0, stream>>>(xb, wqb, bq,  qf);
  gemm_bf16_kernel<<<gg, 256, 0, stream>>>(xb, wkb, bk,  kf);
  gemm_bf16_kernel<<<gg, 256, 0, stream>>>(xb, wvb, bvp, vf);

  // 3) rmsnorm + rope (q, k) ; 4) v transpose
  rmsnorm_rope_kernel<<<kS * kH / 4, 128, 0, stream>>>(qf, gq, freqs, qbn);
  rmsnorm_rope_kernel<<<kS * kH / 4, 128, 0, stream>>>(kf, gk, freqs, kbn);
  vtrans_kernel<<<(int)(SC / 256), 256, 0, stream>>>(vf, vtb);

  // 5) flash attention
  attn_kernel<<<dim3(kS / 16 / ATT_WAVES, kH), ATT_WAVES * 32, 0, stream>>>(
      qbn, kbn, vtb, attnb);

  // 6) output projection -> fp32 d_out
  gemm_bf16_kernel<<<gg, 256, 0, stream>>>(attnb, wob, bo, (float*)d_out);
}
